// get_model_47029891891542
// MI455X (gfx1250) — matrix operand order repaired and checked by reading, not compiled, not run on hardware
//
#include <hip/hip_runtime.h>
#include <hip/hip_bf16.h>
#include <math.h>

typedef __attribute__((ext_vector_type(16))) _Float16 v16h;
typedef __attribute__((ext_vector_type(8)))  _Float16 v8h;
typedef __attribute__((ext_vector_type(8)))  float    v8f;
typedef unsigned int v4u __attribute__((ext_vector_type(4)));
typedef int          v8i __attribute__((ext_vector_type(8)));
typedef int          v4i __attribute__((ext_vector_type(4)));

typedef __attribute__((address_space(3))) _Float16 lds_h;
typedef __attribute__((address_space(3))) v8h     lds_v8h;

#define BN_EPS 1e-5f

// ---------------- weight convert + K-pad to f16 ----------------
__global__ void pad_weights_kernel(const float* __restrict__ w, _Float16* __restrict__ out,
                                   int Cout, int Cin, int Kp) {
  int tid = blockIdx.x * blockDim.x + threadIdx.x;
  if (tid >= Cout * Kp) return;
  int c = tid / Kp, k = tid % Kp;
  float v = (k < Cin) ? w[(size_t)c * Cin + k] : 0.f;
  out[tid] = (_Float16)v;
}

// ---------------- mlp0: [B*N,6] x [32,6]^T, bn, no relu ----------------
__global__ void mlp0_kernel(const float* __restrict__ xyz, const float* __restrict__ w,
                            const float* __restrict__ bias, const float* __restrict__ g,
                            const float* __restrict__ be, _Float16* __restrict__ out,
                            int npts) {
  int tid = blockIdx.x * blockDim.x + threadIdx.x;
  if (tid >= npts * 32) return;
  int p = tid >> 5, o = tid & 31;
  float acc = bias[o];
#pragma unroll
  for (int c = 0; c < 6; ++c) acc += xyz[(size_t)p * 6 + c] * w[o * 6 + c];
  acc = acc * (g[o] * rsqrtf(1.f + BN_EPS)) + be[o];
  out[tid] = (_Float16)acc;
}

// ---------------- farthest point sampling (one block per batch) ----------------
__global__ void fps_kernel(const float* __restrict__ xyz, int stride, int N, int S,
                           float* __restrict__ new_xyz) {
  __shared__ float dist[4096];
  __shared__ float rv[256];
  __shared__ int   ri[256];
  int b = blockIdx.x, t = threadIdx.x;
  const float* base = xyz + (size_t)b * N * stride;
  for (int i = t; i < N; i += 256) dist[i] = 1e10f;
  __syncthreads();
  int far = 0;
  for (int j = 0; j < S; ++j) {
    float fx = base[(size_t)far * stride + 0];
    float fy = base[(size_t)far * stride + 1];
    float fz = base[(size_t)far * stride + 2];
    if (t == 0) {
      new_xyz[((size_t)b * S + j) * 3 + 0] = fx;
      new_xyz[((size_t)b * S + j) * 3 + 1] = fy;
      new_xyz[((size_t)b * S + j) * 3 + 2] = fz;
    }
    float bv = -1.f; int bi = 0;
    for (int i = t; i < N; i += 256) {
      float dx = base[(size_t)i * stride + 0] - fx;
      float dy = base[(size_t)i * stride + 1] - fy;
      float dz = base[(size_t)i * stride + 2] - fz;
      float d  = dx * dx + dy * dy + dz * dz;
      float nd = fminf(dist[i], d);
      dist[i] = nd;
      if (nd > bv) { bv = nd; bi = i; }
    }
    rv[t] = bv; ri[t] = bi;
    __syncthreads();
    for (int s = 128; s > 0; s >>= 1) {
      if (t < s) {
        if (rv[t + s] > rv[t] || (rv[t + s] == rv[t] && ri[t + s] < ri[t])) {
          rv[t] = rv[t + s]; ri[t] = ri[t + s];
        }
      }
      __syncthreads();
    }
    far = ri[0];
    __syncthreads();
  }
}

// ---------------- ball query: one wave per (b,s), ballot compaction ----------------
__global__ void ball_query_kernel(const float* __restrict__ src, int stride, int N,
                                  const float* __restrict__ newx, int S, int ns, float r2,
                                  int* __restrict__ gidx, int B) {
  int wid  = blockIdx.x * (blockDim.x >> 5) + (threadIdx.x >> 5);
  int lane = threadIdx.x & 31;
  if (wid >= B * S) return;
  int b = wid / S, s = wid % S;
  const float* base = src + (size_t)b * N * stride;
  float cx = newx[((size_t)b * S + s) * 3 + 0];
  float cy = newx[((size_t)b * S + s) * 3 + 1];
  float cz = newx[((size_t)b * S + s) * 3 + 2];
  int* out = gidx + ((size_t)b * S + s) * ns;
  int count = 0, first = 0;
  for (int bs = 0; bs < N && count < ns; bs += 32) {
    int i = bs + lane;
    bool in = false;
    if (i < N) {
      float dx = base[(size_t)i * stride + 0] - cx;
      float dy = base[(size_t)i * stride + 1] - cy;
      float dz = base[(size_t)i * stride + 2] - cz;
      in = (dx * dx + dy * dy + dz * dz) <= r2;
    }
    unsigned mask = (unsigned)__ballot(in);
    if (count == 0 && mask != 0) first = bs + __ffs(mask) - 1;
    int pos = count + __popc(mask & ((1u << lane) - 1u));
    if (in && pos < ns) out[pos] = i;
    count += __popc(mask);
  }
  if (count > ns) count = ns;
  for (int p = count + lane; p < ns; p += 32) out[p] = first;
}

// ---------------- build grouped features (centered xyz + gathered feats), per batch ----------------
__global__ void build_grouped_kernel(const float* __restrict__ xyz_src, int src_stride, int Nsrc,
                                     const float* __restrict__ new_xyz,
                                     const _Float16* __restrict__ feats, int Cf,
                                     const int* __restrict__ gidx,
                                     int S, int ns, int Kp, _Float16* __restrict__ Xg, int b) {
  int tid = blockIdx.x * blockDim.x + threadIdx.x;
  if (tid >= S * ns * Kp) return;
  int r = tid / Kp, c = tid % Kp;
  int s = r / ns, k = r % ns;
  int pt = gidx[((size_t)b * S + s) * ns + k];
  if (c < 3) {
    float v = xyz_src[((size_t)b * Nsrc + pt) * src_stride + c]
            - new_xyz[((size_t)b * S + s) * 3 + c];
    Xg[tid] = (_Float16)v;
  } else if (c < 3 + Cf) {
    Xg[tid] = feats[((size_t)b * Nsrc + pt) * Cf + (c - 3)];
  } else {
    Xg[tid] = (_Float16)0.f;
  }
}

// ---------------- sa3 grouped: concat(l2_xyz, l2_points) (all batches, no centering) ----------------
__global__ void build_grouped3_kernel(const float* __restrict__ l2x,
                                      const _Float16* __restrict__ l2p,
                                      _Float16* __restrict__ Xg, int Kp) {
  int tid = blockIdx.x * blockDim.x + threadIdx.x;
  int total = 16 * 256 * Kp;
  if (tid >= total) return;
  int r = tid / Kp, c = tid % Kp; // r = b*256+p
  if (c < 3)            Xg[tid] = (_Float16)l2x[(size_t)r * 3 + c];
  else if (c < 3 + 256) Xg[tid] = l2p[(size_t)r * 256 + (c - 3)];
  else                  Xg[tid] = (_Float16)0.f;
}

// ---------------- WMMA GEMM: Y = relu(bn(X[rows,K] * W[Cout,K]^T + b)) ----------------
// Block = 4 waves sharing one 16-row M tile; the M tile (16 x K f16, contiguous in memory)
// is staged into LDS once per block via the Tensor Data Mover (tensor_load_to_lds,
// TENSORcnt). Each wave computes one 16x16 tile, reading A fragments from LDS
// (ds_load_b128, addrspace(3)) and W fragments from global (L2-resident), using two
// independent WMMA accumulation chains to keep the XDL pipeline full.
// rows, Cout multiples of 16; Cout multiple of 64; K multiple of 32, 64 <= K <= 1024.
__global__ void wmma_gemm_bn_relu(const _Float16* __restrict__ X, const _Float16* __restrict__ W,
                                  const float* __restrict__ bias, const float* __restrict__ g,
                                  const float* __restrict__ be, _Float16* __restrict__ Y,
                                  int rows, int K, int Cout) {
  __shared__ _Float16 sX[16 * 1024];   // one 16 x K A-tile (max 32KB)
  int wave = threadIdx.x >> 5;
  int lane = threadIdx.x & 31;
  int tilesN = Cout >> 4;              // multiple of 4
  int nTnG = tilesN >> 2;
  int tm = blockIdx.x / nTnG;
  int tn = (blockIdx.x % nTnG) * 4 + wave;

  _Float16* sxp = sX;                  // provenance handle for the TDM-written tile

  // ---- stage A tile into LDS ----
#if __has_builtin(__builtin_amdgcn_tensor_load_to_lds)
  if (wave == 0) {
    unsigned long long gaddr =
        (unsigned long long)X + (unsigned long long)tm * 16 * K * 2;
    // D# group0: count=1 (valid user descriptor), lds_addr=0, global_addr, type=2
    v4u g0;
    g0[0] = 1u;
    g0[1] = 0u;
    g0[2] = (unsigned)(gaddr & 0xffffffffu);
    g0[3] = (unsigned)((gaddr >> 32) & 0x01ffffffu) | (2u << 30);
    // D# group1: wg_mask=0, data_size=1 (2B), tensor_dim0=K, tensor_dim1=16,
    //            tile_dim0=K, tile_dim1=16, tensor_dim0_stride=K
    v8i g1;
    g1[0] = 1 << 16;                                   // data_size = 2 bytes
    g1[1] = (K & 0xffff) << 16;                        // tensor_dim0[15:0]
    g1[2] = ((K >> 16) & 0xffff) | (16 << 16);         // tensor_dim0[31:16] | tensor_dim1[15:0]
    g1[3] = (K & 0xffff) << 16;                        // tensor_dim1[31:16]=0 | tile_dim0=K
    g1[4] = 16;                                        // tile_dim1=16, tile_dim2=0
    g1[5] = K;                                         // tensor_dim0_stride[31:0]
    g1[6] = 0;
    g1[7] = 0;
    v4i z4 = {0, 0, 0, 0};
#if __clang_major__ >= 23
    v8i z8 = {0, 0, 0, 0, 0, 0, 0, 0};
    __builtin_amdgcn_tensor_load_to_lds(g0, g1, z4, z4, z8, 0);
#else
    __builtin_amdgcn_tensor_load_to_lds(g0, g1, z4, z4, 0);
#endif
    __builtin_amdgcn_s_wait_tensorcnt(0);
  }
#else
  {
    const v8h* src = (const v8h*)(X + (size_t)tm * 16 * K);
    v8h* dst = (v8h*)sX;
    int nvec = (16 * K) >> 3;
    for (int i = threadIdx.x; i < nvec; i += blockDim.x) dst[i] = src[i];
  }
#endif
  // The compiler cannot see that the TDM engine writes sX; route the tile pointer
  // through an asm with a memory clobber so the subsequent LDS reads are real.
  asm volatile("" : "+v"(sxp) : : "memory");
  __syncthreads();

  // Recover the 32-bit DS offset (the low 32 bits of the flat LDS-aperture address)
  // so the A-fragment reads are true ds_load_b128 and stay off the VMEM issue path.
  const lds_h* lsx = (const lds_h*)(unsigned)(size_t)sxp;

  int m16 = lane & 15, half = lane >> 4;
  // A fragment (16-bit A 16x32 layout): row m16, K chunks half*8 and 16+half*8
  const lds_h* arow = lsx + (size_t)m16 * K + half * 8;
  // B fragment (16-bit B 32x16 layout): lane holds col (row of W) m16; K chunks half*8 and 16+half*8
  const _Float16* brow = W + (size_t)(tn * 16 + m16) * K + half * 8;

  auto step = [&](int kb, v8f cacc) -> v8f {
    v8h alo = *(const lds_v8h*)(arow + kb);
    v8h ahi = *(const lds_v8h*)(arow + kb + 16);
    v16h afrag, bfrag;
    v8h blo = *(const v8h*)(brow + kb);
    v8h bhi = *(const v8h*)(brow + kb + 16);
#pragma unroll
    for (int i = 0; i < 8; ++i) { afrag[i] = alo[i]; afrag[i + 8] = ahi[i]; bfrag[i] = blo[i]; bfrag[i + 8] = bhi[i]; }
    return __builtin_amdgcn_wmma_f32_16x16x32_f16(
        /*neg_a=*/false, afrag, /*neg_b=*/false, bfrag,
        /*c_mod=*/(short)0, cacc, /*reuse_a=*/false, /*reuse_b=*/false);
  };

  // Two independent accumulation chains over interleaved K chunks.
  v8f acc0 = {}, acc1 = {};
  int kb = 0;
  for (; kb + 64 <= K; kb += 64) {
    acc0 = step(kb, acc0);
    acc1 = step(kb + 32, acc1);
  }
  if (kb < K) acc0 = step(kb, acc0);   // tail for K % 64 == 32 (160, 288)

  // C/D layout: lane L, VGPR v -> M = v + 8*(L>=16), N = L%16
  int n   = lane & 15;
  int mo  = half * 8;
  int col = tn * 16 + n;
  float scale = g[col] * rsqrtf(1.f + BN_EPS);
  float bv = bias[col], bev = be[col];
#pragma unroll
  for (int v = 0; v < 8; ++v) {
    float val = acc0[v] + acc1[v] + bv;
    val = val * scale + bev;
    val = fmaxf(val, 0.f);
    Y[(size_t)(tm * 16 + mo + v) * Cout + col] = (_Float16)val;
  }
}

// ---------------- max pool over ns rows per group ----------------
__global__ void pool_max_kernel(const _Float16* __restrict__ Y, _Float16* __restrict__ out,
                                int S, int ns, int C, long long out_row0) {
  int tid = blockIdx.x * blockDim.x + threadIdx.x;
  if (tid >= S * C) return;
  int s = tid / C, c = tid % C;
  float m = -1e30f;
  for (int k = 0; k < ns; ++k)
    m = fmaxf(m, (float)Y[((size_t)s * ns + k) * C + c]);
  out[(size_t)(out_row0 + s) * C + c] = (_Float16)m;
}

// ---------------- fc3 + log_softmax (one block per batch row) ----------------
__global__ void fc3_logsoftmax_kernel(const _Float16* __restrict__ h, const float* __restrict__ w,
                                      const float* __restrict__ bias, float* __restrict__ out) {
  __shared__ float logits[40];
  __shared__ float stats[2];
  int b = blockIdx.x, t = threadIdx.x;
  if (t < 40) {
    float acc = bias[t];
    for (int k = 0; k < 256; ++k) acc += (float)h[b * 256 + k] * w[t * 256 + k];
    logits[t] = acc;
  }
  __syncthreads();
  if (t == 0) {
    float mx = logits[0];
    for (int i = 1; i < 40; ++i) mx = fmaxf(mx, logits[i]);
    float sum = 0.f;
    for (int i = 0; i < 40; ++i) sum += expf(logits[i] - mx);
    stats[0] = mx; stats[1] = logf(sum);
  }
  __syncthreads();
  if (t < 40) out[b * 40 + t] = logits[t] - stats[0] - stats[1];
}

// =====================================================================
extern "C" void kernel_launch(void* const* d_in, const int* in_sizes, int n_in,
                              void* d_out, int out_size, void* d_ws, size_t ws_size,
                              hipStream_t stream) {
  (void)in_sizes; (void)n_in; (void)out_size; (void)ws_size;
  const int B = 16, N1 = 4096;
  const int S1 = 512, NS1 = 32;
  const int S2 = 256, NS2 = 64;

  // WMMA layer table: {input index of w, Cout, Cin, Kpadded}
  struct LDesc { int idx, Cout, Cin, Kp; };
  const LDesc L[11] = {
      {5,   64,   35,   64}, {9,   64,   64,   64}, {13, 128,  64,   64},  // sa1
      {17, 128,  131,  160}, {21, 128, 128,  128}, {25, 256, 128,  128},  // sa2
      {29, 256,  259,  288}, {33, 512, 256,  256}, {37, 1024, 512, 512},  // sa3
      {41, 512, 1024, 1024}, {45, 256, 512,  512}                          // fc1, fc2
  };

  const float* xyz = (const float*)d_in[0];

  // ---- workspace layout ----
  char* ws = (char*)d_ws;
  size_t off = 0;
  auto alloc = [&](size_t bytes) -> char* {
    char* p = ws + off;
    off += (bytes + 255) & ~(size_t)255;
    return p;
  };
  _Float16* wp[11];
  for (int i = 0; i < 11; ++i) wp[i] = (_Float16*)alloc((size_t)L[i].Cout * L[i].Kp * 2);
  _Float16* l0f   = (_Float16*)alloc((size_t)B * N1 * 32 * 2);
  float*    l1x   = (float*)   alloc((size_t)B * S1 * 3 * 4);
  float*    l2x   = (float*)   alloc((size_t)B * S2 * 3 * 4);
  int*      gidx1 = (int*)     alloc((size_t)B * S1 * NS1 * 4);
  int*      gidx2 = (int*)     alloc((size_t)B * S2 * NS2 * 4);
  _Float16* Xg    = (_Float16*)alloc((size_t)16384 * 160 * 2);   // also fits 4096x288 for sa3
  _Float16* actA  = (_Float16*)alloc((size_t)4194304 * 2);       // max(16384x256, 4096x1024)
  _Float16* actB  = (_Float16*)alloc((size_t)4194304 * 2);
  _Float16* l1p   = (_Float16*)alloc((size_t)B * S1 * 128 * 2);
  _Float16* l2p   = (_Float16*)alloc((size_t)B * S2 * 256 * 2);
  _Float16* l3    = (_Float16*)alloc((size_t)B * 1024 * 2);
  _Float16* h1    = (_Float16*)alloc((size_t)B * 512 * 2);
  _Float16* h2    = (_Float16*)alloc((size_t)B * 256 * 2);

  auto gemm = [&](const _Float16* X, int li, _Float16* Y, int rows, int Cout) {
    int K = L[li].Kp;
    int blocks = (rows >> 4) * (Cout >> 6);   // tilesM * (tilesN/4)
    wmma_gemm_bn_relu<<<blocks, 128, 0, stream>>>(
        X, wp[li],
        (const float*)d_in[L[li].idx + 1], (const float*)d_in[L[li].idx + 2],
        (const float*)d_in[L[li].idx + 3], Y, rows, K, Cout);
  };

  // ---- weight prep ----
  for (int i = 0; i < 11; ++i) {
    int total = L[i].Cout * L[i].Kp;
    pad_weights_kernel<<<(total + 255) / 256, 256, 0, stream>>>(
        (const float*)d_in[L[i].idx], wp[i], L[i].Cout, L[i].Cin, L[i].Kp);
  }

  // ---- mlp0 ----
  mlp0_kernel<<<(B * N1 * 32 + 255) / 256, 256, 0, stream>>>(
      xyz, (const float*)d_in[1], (const float*)d_in[2],
      (const float*)d_in[3], (const float*)d_in[4], l0f, B * N1);

  // ---- SA1 sampling/grouping ----
  fps_kernel<<<B, 256, 0, stream>>>(xyz, 6, N1, S1, l1x);
  ball_query_kernel<<<(B * S1 + 3) / 4, 128, 0, stream>>>(
      xyz, 6, N1, l1x, S1, NS1, 0.2f * 0.2f, gidx1, B);

  for (int b = 0; b < B; ++b) {
    int rows = S1 * NS1; // 16384
    int tot = rows * 64;
    build_grouped_kernel<<<(tot + 255) / 256, 256, 0, stream>>>(
        xyz, 6, N1, l1x, l0f, 32, gidx1, S1, NS1, 64, Xg, b);
    gemm(Xg,   0, actA, rows, 64);
    gemm(actA, 1, actB, rows, 64);
    gemm(actB, 2, actA, rows, 128);
    pool_max_kernel<<<(S1 * 128 + 255) / 256, 256, 0, stream>>>(
        actA, l1p, S1, NS1, 128, (long long)b * S1);
  }

  // ---- SA2 ----
  fps_kernel<<<B, 256, 0, stream>>>(l1x, 3, S1, S2, l2x);
  ball_query_kernel<<<(B * S2 + 3) / 4, 128, 0, stream>>>(
      l1x, 3, S1, l2x, S2, NS2, 0.4f * 0.4f, gidx2, B);

  for (int b = 0; b < B; ++b) {
    int rows = S2 * NS2; // 16384
    int tot = rows * 160;
    build_grouped_kernel<<<(tot + 255) / 256, 256, 0, stream>>>(
        l1x, 3, S1, l2x, l1p, 128, gidx2, S2, NS2, 160, Xg, b);
    gemm(Xg,   3, actA, rows, 128);
    gemm(actA, 4, actB, rows, 128);
    gemm(actB, 5, actA, rows, 256);
    pool_max_kernel<<<(S2 * 256 + 255) / 256, 256, 0, stream>>>(
        actA, l2p, S2, NS2, 256, (long long)b * S2);
  }

  // ---- SA3 (group_all, all batches at once: 4096 rows) ----
  {
    int tot = B * 256 * 288;
    build_grouped3_kernel<<<(tot + 255) / 256, 256, 0, stream>>>(l2x, l2p, Xg, 288);
    gemm(Xg,   6, actA, 4096, 256);
    gemm(actA, 7, actB, 4096, 512);
    gemm(actB, 8, actA, 4096, 1024);
    for (int b = 0; b < B; ++b)
      pool_max_kernel<<<(1024 + 255) / 256, 256, 0, stream>>>(
          actA + (size_t)b * 256 * 1024, l3, 1, 256, 1024, (long long)b);
  }

  // ---- FC head ----
  gemm(l3, 9,  h1, 16, 512);
  gemm(h1, 10, h2, 16, 256);
  fc3_logsoftmax_kernel<<<B, 64, 0, stream>>>(
      h2, (const float*)d_in[49], (const float*)d_in[50], (float*)d_out);
}
